// GPTMini_9474697855448
// MI455X (gfx1250) — compile-verified, matrix operand order checked by reading, not run on hardware
//
#include <hip/hip_runtime.h>
#include <hip/hip_bf16.h>
#include <math.h>
#include <stdint.h>

#define D_MODEL  1024
#define N_HEAD   16
#define HEAD_DIM 64
#define SEQ_T    1024
#define BATCH    2
#define N_LAYER  10
#define D_FF     4096
#define VOCAB    50257
#define VOCAB_PAD 50304          // 393 * 128
#define MTOT     (BATCH*SEQ_T)   // 2048

typedef __bf16 bf16_t;
typedef __attribute__((ext_vector_type(16))) __bf16 v16bf;
typedef __attribute__((ext_vector_type(4)))  __bf16 v4bf;
typedef __attribute__((ext_vector_type(8)))  float  v8f;
typedef __attribute__((ext_vector_type(4)))  float  v4f;

// ---------------------------------------------------------------------------
// WMMA helpers (CDNA5 wave32, v_wmma_f32_16x16x32_bf16).
// A-frag (16x32, 16-bit): lane&15 = row; elems e<8 -> K = half*8+e,
// e>=8 -> K = 16+half*8+(e-8), half = lane>>4 (ISA 7.12.2).  B-frag is the
// same pattern with N as the per-lane row when the tile is stored [N][K].
// ---------------------------------------------------------------------------
__device__ __forceinline__ v16bf frag_lds(const bf16_t* base, int half) {
  v16bf f;
  const bf16_t* p0 = base + half * 8;
  const bf16_t* p1 = base + 16 + half * 8;
#pragma unroll
  for (int e = 0; e < 8; ++e) { f[e] = p0[e]; f[8 + e] = p1[e]; }
  return f;
}

__device__ __forceinline__ v16bf frag_gf32(const float* base, int half) {
  v16bf f;
  const float* p0 = base + half * 8;
  const float* p1 = base + 16 + half * 8;
#pragma unroll
  for (int e = 0; e < 8; ++e) { f[e] = (bf16_t)p0[e]; f[8 + e] = (bf16_t)p1[e]; }
  return f;
}

__device__ __forceinline__ v8f wmma_bf16(v16bf a, v16bf b, v8f c) {
  return __builtin_amdgcn_wmma_f32_16x16x32_bf16(false, a, false, b, (short)0, c,
                                                 false, false);
}

// Async global->LDS copy (CDNA5, tracked by ASYNCcnt).  Per-lane 16B chunk.
__device__ __forceinline__ void async_load_lds_b128(uint32_t lds_off,
                                                    const void* gaddr) {
  asm volatile("global_load_async_to_lds_b128 %0, %1, off"
               :: "v"(lds_off), "v"(gaddr) : "memory");
}
__device__ __forceinline__ void wait_async0() {
  asm volatile("s_wait_asynccnt 0" ::: "memory");
}

// ---------------------------------------------------------------------------
// Embedding: h[row] = emb[x[row]] + pos[row % T]
// ---------------------------------------------------------------------------
__global__ __launch_bounds__(256) void embed_kernel(const int* __restrict__ x,
    const float* __restrict__ emb, const float* __restrict__ pos,
    float* __restrict__ h) {
  const int row = blockIdx.x;
  const int tid = threadIdx.x;
  const int tok = x[row];
  const int t = row & (SEQ_T - 1);
#pragma unroll
  for (int i = 0; i < 4; ++i) {
    int d = tid * 4 + i;
    h[(size_t)row * D_MODEL + d] =
        emb[(size_t)tok * D_MODEL + d] + pos[(size_t)t * D_MODEL + d];
  }
}

// ---------------------------------------------------------------------------
// Weight convert+transpose: W fp32 [K,N] -> WT bf16 [Npad,K] (pad rows = 0)
// ---------------------------------------------------------------------------
__global__ __launch_bounds__(256) void convT_kernel(const float* __restrict__ W,
    bf16_t* __restrict__ WT, int K, int N) {
  __shared__ float tile[32][33];
  const int kb = blockIdx.y * 32;
  const int nb = blockIdx.x * 32;
  const int r  = threadIdx.x >> 3;         // 0..31
  const int c0 = (threadIdx.x & 7) * 4;    // 0..28
#pragma unroll
  for (int j = 0; j < 4; ++j) {
    int n = nb + c0 + j;
    tile[r][c0 + j] = (n < N) ? W[(size_t)(kb + r) * N + n] : 0.f;
  }
  __syncthreads();
  v4bf o;
#pragma unroll
  for (int j = 0; j < 4; ++j) o[j] = (bf16_t)tile[c0 + j][r];
  *(v4bf*)&WT[(size_t)(nb + r) * K + kb + c0] = o;
}

// Pack one layer's q/k/v biases into a contiguous 3072-vector
__global__ __launch_bounds__(256) void pack_bias_kernel(
    const float* __restrict__ bq, const float* __restrict__ bk,
    const float* __restrict__ bv, float* __restrict__ b3) {
  int i = blockIdx.x * 256 + threadIdx.x;   // 0..3071
  float v;
  if (i < 1024)       v = bq[i];
  else if (i < 2048)  v = bk[i - 1024];
  else                v = bv[i - 2048];
  b3[i] = v;
}

// ---------------------------------------------------------------------------
// Pipelined GEMM (bf16 pre-transposed weights WT[Npad][K]):
//   C[M,N] = A[M,K] @ WT^T + bias  (mode 1: exact GELU)
// Double-buffered LDS; W tiles staged with async global->LDS b128 copies
// overlapped with WMMA compute; A tiles staged via registers (fp32->bf16).
// Block 256 threads / 8 waves, tile 128x128x32.
// ---------------------------------------------------------------------------
__device__ __forceinline__ void loadA_tile(const float* __restrict__ A, int m0,
                                           int K, int k0, int tid, v4f* areg) {
#pragma unroll
  for (int i = 0; i < 4; ++i) {
    int idx4 = tid + i * 256;              // 0..1023
    int row = idx4 >> 3, col = (idx4 & 7) * 4;
    areg[i] = *(const v4f*)&A[(size_t)(m0 + row) * K + k0 + col];
  }
}

__device__ __forceinline__ void storeA_tile(bf16_t* As, int tid,
                                            const v4f* areg) {
#pragma unroll
  for (int i = 0; i < 4; ++i) {
    int idx4 = tid + i * 256;
    int row = idx4 >> 3, col = (idx4 & 7) * 4;
    v4bf ab;
#pragma unroll
    for (int j = 0; j < 4; ++j) ab[j] = (bf16_t)areg[i][j];
    *(v4bf*)&As[row * 32 + col] = ab;
  }
}

__device__ __forceinline__ void issueW_tile(const bf16_t* __restrict__ WT,
                                            int n0, int K, int k0, int tid,
                                            uint32_t lds_base) {
#pragma unroll
  for (int i = 0; i < 2; ++i) {
    int c = tid + i * 256;                 // 0..511 16B chunks
    const bf16_t* g = WT + (size_t)(n0 + (c >> 2)) * K + k0 + (c & 3) * 8;
    async_load_lds_b128(lds_base + c * 16, (const void*)g);
  }
}

__global__ __launch_bounds__(256) void gemm_bf16w_kernel(
    const float* __restrict__ A, const bf16_t* __restrict__ WT,
    const float* __restrict__ bias, float* __restrict__ C,
    int M, int N, int K, int mode) {
  __shared__ __align__(16) bf16_t As[2][128 * 32];   // [buf][m][k]
  __shared__ __align__(16) bf16_t Wt[2][128 * 32];   // [buf][n][k]
  const int tid  = threadIdx.x;
  const int m0   = blockIdx.y * 128;
  const int n0   = blockIdx.x * 128;
  const int wave = tid >> 5, lane = tid & 31;
  const int half = lane >> 4, lr = lane & 15;
  const int wm = (wave >> 1) * 32;
  const int wn = (wave & 1) * 64;
  const uint32_t wt_base0 = (uint32_t)(uintptr_t)&Wt[0][0];
  const uint32_t wt_base1 = (uint32_t)(uintptr_t)&Wt[1][0];

  v8f acc[2][4];
#pragma unroll
  for (int i = 0; i < 2; ++i)
#pragma unroll
    for (int j = 0; j < 4; ++j)
#pragma unroll
      for (int r = 0; r < 8; ++r) acc[i][j][r] = 0.f;

  // ---- prologue: stage tile 0 into buffer 0 ----
  v4f areg[4];
  loadA_tile(A, m0, K, 0, tid, areg);
  issueW_tile(WT, n0, K, 0, tid, wt_base0);
  storeA_tile(As[0], tid, areg);
  wait_async0();
  __syncthreads();

  const int nk = K >> 5;
  for (int t = 0; t < nk; ++t) {
    const int cur = t & 1;
    const int nxt = cur ^ 1;
    const bool more = (t + 1) < nk;
    // prefetch next tile: A -> regs, W -> async LDS (other buffer)
    if (more) {
      loadA_tile(A, m0, K, (t + 1) * 32, tid, areg);
      issueW_tile(WT, n0, K, (t + 1) * 32, tid, cur ? wt_base0 : wt_base1);
    }
    // compute current tile
    v16bf af[2], bfr[4];
#pragma unroll
    for (int i = 0; i < 2; ++i)
      af[i] = frag_lds(&As[cur][(wm + i * 16 + lr) * 32], half);
#pragma unroll
    for (int j = 0; j < 4; ++j)
      bfr[j] = frag_lds(&Wt[cur][(wn + j * 16 + lr) * 32], half);
#pragma unroll
    for (int i = 0; i < 2; ++i)
#pragma unroll
      for (int j = 0; j < 4; ++j)
        acc[i][j] = wmma_bf16(af[i], bfr[j], acc[i][j]);
    // finish staging next tile
    if (more) {
      storeA_tile(As[nxt], tid, areg);
      wait_async0();
    }
    __syncthreads();
  }

#pragma unroll
  for (int i = 0; i < 2; ++i)
#pragma unroll
    for (int j = 0; j < 4; ++j) {
      int col = n0 + wn + j * 16 + lr;
      if (col >= N) continue;
      float bia = bias[col];
#pragma unroll
      for (int r = 0; r < 8; ++r) {
        int row = m0 + wm + i * 16 + half * 8 + r;
        float v = acc[i][j][r] + bia;
        if (mode == 1)
          v = 0.5f * v * (1.0f + erff(v * 0.70710678118654752f));
        C[(size_t)row * N + col] = v;
      }
    }
}

// ---------------------------------------------------------------------------
// Fallback GEMM (fp32 weights [K,N]); uniform tile-level bounds branch.
// ---------------------------------------------------------------------------
__global__ __launch_bounds__(256) void gemm_f32w_kernel(
    const float* __restrict__ A, const float* __restrict__ W,
    const float* __restrict__ bias, float* __restrict__ C,
    int M, int N, int K, int mode) {
  __shared__ __align__(16) bf16_t As[128 * 32];
  __shared__ __align__(16) bf16_t Wt[128 * 32];
  const int tid  = threadIdx.x;
  const int m0   = blockIdx.y * 128;
  const int n0   = blockIdx.x * 128;
  const int wave = tid >> 5, lane = tid & 31;
  const int half = lane >> 4, lr = lane & 15;
  const int wm = (wave >> 1) * 32;
  const int wn = (wave & 1) * 64;
  const bool fastN = ((N & 3) == 0) && (n0 + 128 <= N);

  v8f acc[2][4];
#pragma unroll
  for (int i = 0; i < 2; ++i)
#pragma unroll
    for (int j = 0; j < 4; ++j)
#pragma unroll
      for (int r = 0; r < 8; ++r) acc[i][j][r] = 0.f;

  for (int k0 = 0; k0 < K; k0 += 32) {
    __syncthreads();
#pragma unroll
    for (int i = 0; i < 4; ++i) {
      int idx4 = tid + i * 256;
      int row = idx4 >> 3, col = (idx4 & 7) * 4;
      v4f a4 = *(const v4f*)&A[(size_t)(m0 + row) * K + k0 + col];
      v4bf ab;
#pragma unroll
      for (int j = 0; j < 4; ++j) ab[j] = (bf16_t)a4[j];
      *(v4bf*)&As[row * 32 + col] = ab;
    }
    if (fastN) {
#pragma unroll
      for (int i = 0; i < 4; ++i) {
        int idx4 = tid + i * 256;
        int wr = idx4 >> 5;                  // k row 0..31
        int wc = (idx4 & 31) * 4;            // n col 0..124
        v4f w4 = *(const v4f*)&W[(size_t)(k0 + wr) * N + n0 + wc];
#pragma unroll
        for (int j = 0; j < 4; ++j) Wt[(wc + j) * 32 + wr] = (bf16_t)w4[j];
      }
    } else {
#pragma unroll
      for (int i = 0; i < 16; ++i) {
        int idx = tid + i * 256;
        int wc = idx & 127, wr = idx >> 7;
        int gn = n0 + wc;
        Wt[wc * 32 + wr] =
            (gn < N) ? (bf16_t)W[(size_t)(k0 + wr) * N + gn] : (bf16_t)0.f;
      }
    }
    __syncthreads();

    v16bf af[2], bfr[4];
#pragma unroll
    for (int i = 0; i < 2; ++i)
      af[i] = frag_lds(&As[(wm + i * 16 + lr) * 32], half);
#pragma unroll
    for (int j = 0; j < 4; ++j)
      bfr[j] = frag_lds(&Wt[(wn + j * 16 + lr) * 32], half);
#pragma unroll
    for (int i = 0; i < 2; ++i)
#pragma unroll
      for (int j = 0; j < 4; ++j)
        acc[i][j] = wmma_bf16(af[i], bfr[j], acc[i][j]);
  }

#pragma unroll
  for (int i = 0; i < 2; ++i)
#pragma unroll
    for (int j = 0; j < 4; ++j) {
      int col = n0 + wn + j * 16 + lr;
      if (col >= N) continue;
      float bia = bias[col];
#pragma unroll
      for (int r = 0; r < 8; ++r) {
        int row = m0 + wm + i * 16 + half * 8 + r;
        float v = acc[i][j][r] + bia;
        if (mode == 1)
          v = 0.5f * v * (1.0f + erff(v * 0.70710678118654752f));
        C[(size_t)row * N + col] = v;
      }
    }
}

// ---------------------------------------------------------------------------
// Flash attention: block = (batch, head, 64 query rows), 4 waves x 16 rows.
// ld = row stride of q/k/v (3072 for fused QKV buffer, 1024 in fallback).
// ---------------------------------------------------------------------------
__global__ __launch_bounds__(128) void attn_kernel(const float* __restrict__ q,
    const float* __restrict__ k, const float* __restrict__ v,
    float* __restrict__ o, int ld) {
  __shared__ bf16_t Ks[32 * 64];      // [time][hd]
  __shared__ bf16_t Vt[64 * 32];      // [hd][time]
  __shared__ bf16_t Ps[4 * 16 * 32];  // per-wave P tile [m][k]

  const int tid  = threadIdx.x;
  const int wave = tid >> 5, lane = tid & 31;
  const int half = lane >> 4, lr = lane & 15;
  const int bid = blockIdx.x;
  const int qb = bid & 15;
  const int hh = (bid >> 4) & 15;
  const int bb = bid >> 8;

  const int qrow = qb * 64 + wave * 16 + lr;
  const float* qbase = q + (size_t)(bb * SEQ_T + qrow) * ld + hh * HEAD_DIM;
  v16bf qf0 = frag_gf32(qbase, half);
  v16bf qf1 = frag_gf32(qbase + 32, half);

  float mi[8], li[8];
  v8f accO[4];
#pragma unroll
  for (int r = 0; r < 8; ++r) { mi[r] = -3.0e38f; li[r] = 0.f; }
#pragma unroll
  for (int t2 = 0; t2 < 4; ++t2)
#pragma unroll
    for (int r = 0; r < 8; ++r) accO[t2][r] = 0.f;

  bf16_t* Pw = &Ps[wave * 16 * 32];
  const int nkb = qb * 2 + 2;

  for (int kb2 = 0; kb2 < nkb; ++kb2) {
    const int tbase = kb2 * 32;
    __syncthreads();
#pragma unroll
    for (int i = 0; i < 16; ++i) {
      int idx = tid + i * 128;
      int tt = idx >> 6, hd = idx & 63;
      size_t g = (size_t)(bb * SEQ_T + tbase + tt) * ld + hh * HEAD_DIM + hd;
      Ks[tt * 64 + hd] = (bf16_t)k[g];
      Vt[hd * 32 + tt] = (bf16_t)v[g];
    }
    __syncthreads();

    v8f c[2];
#pragma unroll
    for (int nt = 0; nt < 2; ++nt) {
#pragma unroll
      for (int r = 0; r < 8; ++r) c[nt][r] = 0.f;
      const bf16_t* kr = &Ks[(nt * 16 + lr) * 64];
      c[nt] = wmma_bf16(qf0, frag_lds(kr, half), c[nt]);
      c[nt] = wmma_bf16(qf1, frag_lds(kr + 32, half), c[nt]);
    }

#pragma unroll
    for (int nt = 0; nt < 2; ++nt)
#pragma unroll
      for (int r = 0; r < 8; ++r) {
        int rg = qb * 64 + wave * 16 + half * 8 + r;
        int cg = tbase + nt * 16 + lr;
        float s = c[nt][r] * 0.125f;          // 1/sqrt(64)
        c[nt][r] = (cg > rg) ? -3.0e38f : s;  // causal mask
      }

#pragma unroll
    for (int r = 0; r < 8; ++r) {
      float mx = fmaxf(c[0][r], c[1][r]);
      for (int off = 8; off >= 1; off >>= 1)
        mx = fmaxf(mx, __shfl_xor(mx, off, 16));
      float mnew = fmaxf(mi[r], mx);
      float corr = expf(mi[r] - mnew);
      float p0 = expf(c[0][r] - mnew);
      float p1 = expf(c[1][r] - mnew);
      float rs = p0 + p1;
      for (int off = 8; off >= 1; off >>= 1) rs += __shfl_xor(rs, off, 16);
      li[r] = li[r] * corr + rs;
      mi[r] = mnew;
      c[0][r] = p0; c[1][r] = p1;
#pragma unroll
      for (int t2 = 0; t2 < 4; ++t2) accO[t2][r] *= corr;
    }

#pragma unroll
    for (int nt = 0; nt < 2; ++nt)
#pragma unroll
      for (int r = 0; r < 8; ++r)
        Pw[(half * 8 + r) * 32 + nt * 16 + lr] = (bf16_t)c[nt][r];
    __syncthreads();

    v16bf pf = frag_lds(&Pw[lr * 32], half);
#pragma unroll
    for (int t2 = 0; t2 < 4; ++t2)
      accO[t2] = wmma_bf16(pf, frag_lds(&Vt[(t2 * 16 + lr) * 32], half),
                           accO[t2]);
  }

#pragma unroll
  for (int t2 = 0; t2 < 4; ++t2)
#pragma unroll
    for (int r = 0; r < 8; ++r) {
      int rg = qb * 64 + wave * 16 + half * 8 + r;
      int col = hh * HEAD_DIM + t2 * 16 + lr;
      o[(size_t)(bb * SEQ_T + rg) * D_MODEL + col] = accO[t2][r] / li[r];
    }
}

// ---------------------------------------------------------------------------
// Fused residual + LayerNorm (1 row / block)
// ---------------------------------------------------------------------------
__global__ __launch_bounds__(256) void ln_kernel(const float* __restrict__ hin,
    const float* __restrict__ add, const float* __restrict__ g,
    const float* __restrict__ bta, float* __restrict__ hout) {
  __shared__ float red[256];
  const int row = blockIdx.x, tid = threadIdx.x;
  const float* hp = hin + (size_t)row * D_MODEL;
  const bool hasAdd = (add != nullptr);
  float vals[4];
  float s = 0.f;
#pragma unroll
  for (int i = 0; i < 4; ++i) {
    int d = tid * 4 + i;
    float x = hp[d];
    if (hasAdd) x += add[(size_t)row * D_MODEL + d];
    vals[i] = x; s += x;
  }
  red[tid] = s; __syncthreads();
  for (int off = 128; off > 0; off >>= 1) {
    if (tid < off) red[tid] += red[tid + off];
    __syncthreads();
  }
  float mu = red[0] * (1.f / D_MODEL);
  __syncthreads();
  float vs = 0.f;
#pragma unroll
  for (int i = 0; i < 4; ++i) { float d = vals[i] - mu; vs += d * d; }
  red[tid] = vs; __syncthreads();
  for (int off = 128; off > 0; off >>= 1) {
    if (tid < off) red[tid] += red[tid + off];
    __syncthreads();
  }
  float inv = rsqrtf(red[0] * (1.f / D_MODEL) + 1e-5f);
#pragma unroll
  for (int i = 0; i < 4; ++i) {
    int d = tid * 4 + i;
    hout[(size_t)row * D_MODEL + d] = (vals[i] - mu) * inv * g[d] + bta[d];
  }
}

// ---------------------------------------------------------------------------
// Host orchestration
// ---------------------------------------------------------------------------
extern "C" void kernel_launch(void* const* d_in, const int* in_sizes, int n_in,
                              void* d_out, int out_size, void* d_ws,
                              size_t ws_size, hipStream_t stream) {
  const int*   x     = (const int*)  d_in[0];
  const float* emb   = (const float*)d_in[1];
  const float* pos   = (const float*)d_in[2];
  const float* Wq    = (const float*)d_in[3];
  const float* bq    = (const float*)d_in[4];
  const float* Wk    = (const float*)d_in[5];
  const float* bk    = (const float*)d_in[6];
  const float* Wv    = (const float*)d_in[7];
  const float* bv    = (const float*)d_in[8];
  const float* ln_g  = (const float*)d_in[9];
  const float* ln_b  = (const float*)d_in[10];
  const float* W1    = (const float*)d_in[11];
  const float* b1    = (const float*)d_in[12];
  const float* W2    = (const float*)d_in[13];
  const float* b2    = (const float*)d_in[14];
  const float* lnf_g = (const float*)d_in[15];
  const float* lnf_b = (const float*)d_in[16];
  const float* Wout  = (const float*)d_in[17];
  const float* bout  = (const float*)d_in[18];
  float* out = (float*)d_out;

  const size_t F2M = 2097152;            // 2M floats = one [2048,1024] buffer
  float* ws  = (float*)d_ws;
  float* h0  = ws;
  float* h1  = ws + 1 * F2M;
  float* obf = ws + 2 * F2M;
  float* mid = ws + 3 * F2M;             // [2048,4096] = 4*F2M floats
  float* qkv = ws + 7 * F2M;             // [2048,3072] = 3*F2M floats
  float* b3  = ws + 10 * F2M;            // 3072 floats
  const size_t ACT_FLOATS = 10 * F2M + 4096;

  const size_t szQKV  = (size_t)N_LAYER * 3072 * 1024;   // bf16 elems
  const size_t szW1   = (size_t)N_LAYER * D_FF * D_MODEL;
  const size_t szW2   = (size_t)N_LAYER * D_MODEL * D_FF;
  const size_t szWout = (size_t)VOCAB_PAD * D_MODEL;
  const size_t needBytes =
      ACT_FLOATS * 4 + (szQKV + szW1 + szW2 + szWout) * 2;

  embed_kernel<<<MTOT, 256, 0, stream>>>(x, emb, pos, h0);

  if (ws_size >= needBytes) {
    // ---------------- bf16 pre-transposed weight path -------------------
    bf16_t* wreg   = (bf16_t*)((char*)d_ws + ACT_FLOATS * 4);
    bf16_t* WqkvT  = wreg;
    bf16_t* W1T    = WqkvT + szQKV;
    bf16_t* W2T    = W1T + szW1;
    bf16_t* WoutT  = W2T + szW2;

    for (int l = 0; l < N_LAYER; ++l) {
      bf16_t* dst = WqkvT + (size_t)l * 3072 * 1024;
      convT_kernel<<<dim3(32, 32), 256, 0, stream>>>(
          Wq + (size_t)l * D_MODEL * D_MODEL, dst, D_MODEL, D_MODEL);
      convT_kernel<<<dim3(32, 32), 256, 0, stream>>>(
          Wk + (size_t)l * D_MODEL * D_MODEL, dst + (size_t)1024 * 1024,
          D_MODEL, D_MODEL);
      convT_kernel<<<dim3(32, 32), 256, 0, stream>>>(
          Wv + (size_t)l * D_MODEL * D_MODEL, dst + (size_t)2048 * 1024,
          D_MODEL, D_MODEL);
      convT_kernel<<<dim3(128, 32), 256, 0, stream>>>(
          W1 + (size_t)l * D_MODEL * D_FF, W1T + (size_t)l * D_FF * D_MODEL,
          D_MODEL, D_FF);
      convT_kernel<<<dim3(32, 128), 256, 0, stream>>>(
          W2 + (size_t)l * D_FF * D_MODEL, W2T + (size_t)l * D_MODEL * D_FF,
          D_FF, D_MODEL);
    }
    convT_kernel<<<dim3(VOCAB_PAD / 32, 32), 256, 0, stream>>>(
        Wout, WoutT, D_MODEL, VOCAB);

    const dim3 gQKV(24, 16), gFF1(32, 16), gFF2(8, 16), gOut(393, 16);
    for (int l = 0; l < N_LAYER; ++l) {
      const float* g  = ln_g + (size_t)l * D_MODEL;
      const float* bb = ln_b + (size_t)l * D_MODEL;
      pack_bias_kernel<<<12, 256, 0, stream>>>(
          bq + (size_t)l * D_MODEL, bk + (size_t)l * D_MODEL,
          bv + (size_t)l * D_MODEL, b3);
      gemm_bf16w_kernel<<<gQKV, 256, 0, stream>>>(
          h0, WqkvT + (size_t)l * 3072 * 1024, b3, qkv, MTOT, 3072, D_MODEL, 0);
      attn_kernel<<<BATCH * N_HEAD * (SEQ_T / 64), 128, 0, stream>>>(
          qkv, qkv + 1024, qkv + 2048, obf, 3072);
      ln_kernel<<<MTOT, 256, 0, stream>>>(h0, obf, g, bb, h1);
      gemm_bf16w_kernel<<<gFF1, 256, 0, stream>>>(
          h1, W1T + (size_t)l * D_FF * D_MODEL, b1 + (size_t)l * D_FF, mid,
          MTOT, D_FF, D_MODEL, 1);
      gemm_bf16w_kernel<<<gFF2, 256, 0, stream>>>(
          mid, W2T + (size_t)l * D_MODEL * D_FF, b2 + (size_t)l * D_MODEL, obf,
          MTOT, D_MODEL, D_FF, 0);
      ln_kernel<<<MTOT, 256, 0, stream>>>(h1, obf, g, bb, h0);
    }
    ln_kernel<<<MTOT, 256, 0, stream>>>(h0, nullptr, lnf_g, lnf_b, h1);
    gemm_bf16w_kernel<<<gOut, 256, 0, stream>>>(h1, WoutT, bout, out, MTOT,
                                                VOCAB, D_MODEL, 0);
  } else {
    // ---------------- fallback: fp32 weights straight from inputs -------
    float* qbf = qkv;
    float* kbf = qkv + F2M;
    float* vbf = qkv + 2 * F2M;
    const dim3 gQKV(8, 16), gFF1(32, 16), gOut(393, 16);
    for (int l = 0; l < N_LAYER; ++l) {
      const float* g  = ln_g + (size_t)l * D_MODEL;
      const float* bb = ln_b + (size_t)l * D_MODEL;
      gemm_f32w_kernel<<<gQKV, 256, 0, stream>>>(
          h0, Wq + (size_t)l * D_MODEL * D_MODEL, bq + (size_t)l * D_MODEL,
          qbf, MTOT, D_MODEL, D_MODEL, 0);
      gemm_f32w_kernel<<<gQKV, 256, 0, stream>>>(
          h0, Wk + (size_t)l * D_MODEL * D_MODEL, bk + (size_t)l * D_MODEL,
          kbf, MTOT, D_MODEL, D_MODEL, 0);
      gemm_f32w_kernel<<<gQKV, 256, 0, stream>>>(
          h0, Wv + (size_t)l * D_MODEL * D_MODEL, bv + (size_t)l * D_MODEL,
          vbf, MTOT, D_MODEL, D_MODEL, 0);
      attn_kernel<<<BATCH * N_HEAD * (SEQ_T / 64), 128, 0, stream>>>(
          qbf, kbf, vbf, obf, D_MODEL);
      ln_kernel<<<MTOT, 256, 0, stream>>>(h0, obf, g, bb, h1);
      gemm_f32w_kernel<<<gFF1, 256, 0, stream>>>(
          h1, W1 + (size_t)l * D_MODEL * D_FF, b1 + (size_t)l * D_FF, mid,
          MTOT, D_FF, D_MODEL, 1);
      gemm_f32w_kernel<<<gQKV, 256, 0, stream>>>(
          mid, W2 + (size_t)l * D_FF * D_MODEL, b2 + (size_t)l * D_MODEL, obf,
          MTOT, D_MODEL, D_FF, 0);
      ln_kernel<<<MTOT, 256, 0, stream>>>(h1, obf, g, bb, h0);
    }
    ln_kernel<<<MTOT, 256, 0, stream>>>(h0, nullptr, lnf_g, lnf_b, h1);
    gemm_f32w_kernel<<<gOut, 256, 0, stream>>>(h1, Wout, bout, out, MTOT,
                                               VOCAB, D_MODEL, 0);
  }
}
